// Model_2654289789219
// MI455X (gfx1250) — matrix operand order repaired and checked by reading, not compiled, not run on hardware
//
#include <hip/hip_runtime.h>

// ---------------------------------------------------------------------------
// CDNA5 (gfx1250) sparse 3D U-Net, implicit-GEMM with V_WMMA_F32_16X16X32_BF16
// - one wave per 16-voxel tile, all Cout tiles per wave (A-fragment reuse)
// - per-tap weights staged into LDS by the Tensor Data Mover (tensor_load_to_lds,
//   TENSORcnt-tracked), falling back to cooperative ld+ds_store if unavailable
// ---------------------------------------------------------------------------

typedef __bf16 v16bf __attribute__((ext_vector_type(16)));
typedef float  v8f   __attribute__((ext_vector_type(8)));
typedef unsigned int u32x4 __attribute__((ext_vector_type(4)));
typedef int          i32x4 __attribute__((ext_vector_type(4)));
typedef int          i32x8 __attribute__((ext_vector_type(8)));

#if defined(__gfx1250__) && __has_builtin(__builtin_amdgcn_tensor_load_to_lds) && \
    __has_builtin(__builtin_amdgcn_s_wait_tensorcnt)
#define USE_TDM 1
#endif

__device__ __forceinline__ float bf2f(unsigned short u) {
    return __uint_as_float(((unsigned int)u) << 16);
}
__device__ __forceinline__ unsigned short f2bf(float f) {
    unsigned int u = __float_as_uint(f);
    unsigned int r = u + 0x7FFFu + ((u >> 16) & 1u);   // RNE
    return (unsigned short)(r >> 16);
}

union Frag { uint4 q[2]; v16bf v; };

// A fragment (16x32 bf16): lane (m = l&15, half = l>>4).
// element j: K = (j<8 ? j : j+8) + half*8  -> two contiguous 8-channel chunks.
__device__ __forceinline__ v16bf load_a_frag(const unsigned short* base, int k0,
                                             int half, int Cin, bool valid) {
    Frag u;
    int kA = k0 + half * 8;
    int kB = k0 + 16 + half * 8;
    uint4 z4; z4.x = z4.y = z4.z = z4.w = 0u;
    u.q[0] = (valid && kA < Cin) ? *(const uint4*)(base + kA) : z4;
    u.q[1] = (valid && kB < Cin) ? *(const uint4*)(base + kB) : z4;
    return u.v;
}

#ifdef USE_TDM
// 1D TDM copy: `bytes` contiguous bytes global -> LDS (bytes % 8 == 0).
// D# per CDNA5 ISA: group0 {count=1, lds_addr, global_addr,
// type=2}; group1 {mask=0, data_size=8B, tensor_dim0=tile_dim0=n8, dims1=1,
// dim0_stride=n8}; groups 2/3 zero (<=2D tensor).
__device__ __forceinline__ void tdm_load_slice(const unsigned short* gsrc,
                                               unsigned short* lds,
                                               unsigned int bytes) {
    unsigned long long ga = (unsigned long long)(size_t)gsrc;
    unsigned int la = (unsigned int)(size_t)lds;       // flat addr[31:0] = LDS byte addr
    unsigned int n8 = bytes >> 3;                      // 8-byte elements (<= 65535)
    u32x4 g0;
    g0.x = 1u;                                         // count=1, user descriptor
    g0.y = la;                                         // lds_addr
    g0.z = (unsigned int)ga;                           // global_addr[31:0]
    g0.w = (unsigned int)(ga >> 32) | (2u << 30);      // global_addr[56:32] | type=2
    i32x8 g1;
    g1[0] = (int)(3u << 16);                           // workgroup_mask=0, data_size=8B
    g1[1] = (int)((n8 & 0xFFFFu) << 16);               // tensor_dim0[15:0] @bit48
    g1[2] = (int)(((n8 >> 16) & 0xFFFFu) | (1u << 16));// tensor_dim0[31:16], tensor_dim1=1
    g1[3] = (int)(n8 << 16);                           // tile_dim0 @bit112
    g1[4] = 1;                                         // tile_dim1=1, tile_dim2=0
    g1[5] = (int)n8;                                   // tensor_dim0_stride[31:0]
    g1[6] = 0; g1[7] = 0;
    i32x4 z4 = {0, 0, 0, 0};
#if __clang_major__ >= 23
    i32x8 z8 = {0, 0, 0, 0, 0, 0, 0, 0};
    __builtin_amdgcn_tensor_load_to_lds(g0, g1, z4, z4, z8, 0);
#else
    __builtin_amdgcn_tensor_load_to_lds(g0, g1, z4, z4, 0);
#endif
}
#endif

#define MODE_SUB  0
#define MODE_DOWN 1
#define MODE_UP   2

// NCT = Cout/16 (compile-time). Block = 8 waves; wave w owns tile blockIdx*8+w.
template <int NCT>
__global__ __launch_bounds__(256) void spconv_wmma(
    const unsigned short* __restrict__ in,   // [Vin][inCS] bf16
    const unsigned short* __restrict__ wt,   // [ntaps][Cout][Kpad] bf16
    unsigned short* __restrict__ out,        // [Vout][outCS] bf16
    const int* __restrict__ alist, const int* __restrict__ acnt,
    int D, int H, int W,                     // output grid dims
    int Cin, int Cout, int Kpad,
    int inCS, int outCS, int outChOff,
    int mode, int ntaps)
{
    extern __shared__ unsigned short smem[];           // [Cout][Kpad] bf16
    int tid = threadIdx.x;
    int waveId = tid >> 5;
    int lane = tid & 31;
    int nA = *acnt;
    if (blockIdx.x * 8 * 16 >= nA) return;             // uniform whole-block exit

    int tile = blockIdx.x * 8 + waveId;
    int base = tile * 16;
    bool tileLive = base < nA;
    int half = lane >> 4;
    int mrow = lane & 15;
    int myIdx = base + mrow;
    bool rowValid = tileLive && (myIdx < nA);
    int vid = rowValid ? alist[myIdx] : 0;
    int X = vid % W; int t1 = vid / W; int Y = t1 % H; int Z = t1 / H;

    v8f acc[NCT];
#pragma unroll
    for (int ct = 0; ct < NCT; ++ct) acc[ct] = v8f{};

    size_t slice = (size_t)Cout * Kpad;
#ifndef USE_TDM
    int nChunk = (Cout * Kpad) >> 3;                   // 16-byte chunks
#endif

    for (int t = 0; t < ntaps; ++t) {
        __syncthreads();                               // prev tap's ds reads done
        const unsigned short* wsrc = wt + (size_t)t * slice;
#ifdef USE_TDM
        if (tid < 32) {                                // wave 0 drives the DMA
            tdm_load_slice(wsrc, smem, (unsigned int)(slice * 2));
            __builtin_amdgcn_s_wait_tensorcnt(0);      // data landed in LDS
        }
#else
        for (int c = tid; c < nChunk; c += 256) {
            *(uint4*)(smem + (size_t)c * 8) = *(const uint4*)(wsrc + (size_t)c * 8);
            if (t + 1 < ntaps)                          // global_prefetch_b8
                __builtin_prefetch(wsrc + slice + (size_t)c * 8, 0, 1);
        }
#endif
        __syncthreads();                               // weights visible to all waves

        bool valid = rowValid;
        int nv = 0;
        if (mode == MODE_SUB) {
            int dz = t / 9 - 1, dy = (t / 3) % 3 - 1, dx = t % 3 - 1;
            int zz = Z + dz, yy = Y + dy, xx = X + dx;
            valid = valid && (unsigned)zz < (unsigned)D && (unsigned)yy < (unsigned)H &&
                    (unsigned)xx < (unsigned)W;
            nv = valid ? (zz * H + yy) * W + xx : 0;
        } else if (mode == MODE_DOWN) {
            int tz = t >> 2, ty = (t >> 1) & 1, tx = t & 1;
            nv = ((Z * 2 + tz) * (H * 2) + (Y * 2 + ty)) * (W * 2) + (X * 2 + tx);
        } else { // MODE_UP: fine-voxel parity selects the weight slice
            int par = ((Z & 1) << 2) | ((Y & 1) << 1) | (X & 1);
            valid = valid && (par == t);
            nv = ((Z >> 1) * (H >> 1) + (Y >> 1)) * (W >> 1) + (X >> 1);
        }
        const unsigned short* aBase = in + (size_t)nv * inCS;

        for (int k0 = 0; k0 < Kpad; k0 += 32) {
            v16bf a = load_a_frag(aBase, k0, half, Cin, valid);   // loaded once,
#pragma unroll
            for (int ct = 0; ct < NCT; ++ct) {                    // reused NCT times
                // B fragment from LDS: lane n holds K = (j<8 ? j : j+8) + half*8 + k0
                const unsigned short* bp =
                    smem + ((size_t)(ct * 16 + (lane & 15))) * Kpad + k0 + half * 8;
                Frag ub;
                ub.q[0] = *(const uint4*)(bp);
                ub.q[1] = *(const uint4*)(bp + 16);
                acc[ct] = __builtin_amdgcn_wmma_f32_16x16x32_bf16(
                    false, a, false, ub.v, (short)0, acc[ct], false, false);
            }
        }
    }
    if (!tileLive) return;
    // D layout: lane n = lane&15, VGPR r -> row M = r + half*8
#pragma unroll
    for (int ct = 0; ct < NCT; ++ct) {
        int coutCh = outChOff + ct * 16 + (lane & 15);
#pragma unroll
        for (int r = 0; r < 8; ++r) {
            int idx2 = base + r + half * 8;
            if (idx2 < nA) {
                int v2 = alist[idx2];
                out[(size_t)v2 * outCS + coutCh] = f2bf(acc[ct][r]);
            }
        }
    }
}

// --------------------------- support kernels -------------------------------

__global__ void mask_init(const int* __restrict__ m, unsigned char* __restrict__ o, int V) {
    int i = blockIdx.x * blockDim.x + threadIdx.x;
    if (i < V) o[i] = m[i] ? 1 : 0;
}

__global__ void mask_down(const unsigned char* __restrict__ fine,
                          unsigned char* __restrict__ coarse, int S) {
    int i = blockIdx.x * blockDim.x + threadIdx.x;
    int V = S * S * S;
    if (i >= V) return;
    int x = i % S, y = (i / S) % S, z = i / (S * S);
    int fS = 2 * S;
    unsigned char r = 0;
    for (int t = 0; t < 8; ++t) {
        int tz = t >> 2, ty = (t >> 1) & 1, tx = t & 1;
        r |= fine[((2 * z + tz) * fS + (2 * y + ty)) * fS + (2 * x + tx)];
    }
    coarse[i] = r;
}

__global__ void compact_kernel(const unsigned char* __restrict__ m, int* __restrict__ list,
                               int* __restrict__ cnt, int V) {
    int i = blockIdx.x * blockDim.x + threadIdx.x;
    if (i < V && m[i]) {
        int p = atomicAdd(cnt, 1);
        list[p] = i;
    }
}

// fp32 conv weights -> bf16 [tap][Cout][Kpad] (K zero-padded to 32)
__global__ void wxform(const float* __restrict__ src, unsigned short* __restrict__ dst,
                       int Cout, int Cin, int ntaps, int Kpad, int coMajor) {
    int gid = blockIdx.x * blockDim.x + threadIdx.x;
    int total = ntaps * Cout * Kpad;
    if (gid >= total) return;
    int k = gid % Kpad; int tmp = gid / Kpad; int co = tmp % Cout; int t = tmp / Cout;
    float v = 0.f;
    if (k < Cin)
        v = coMajor ? src[((size_t)co * Cin + k) * ntaps + t]
                    : src[((size_t)k * Cout + co) * ntaps + t];
    dst[gid] = f2bf(v);
}

// Stem conv: 3 -> 16 channels, 3^3, input masked (x*m0)
__global__ void conv0_kernel(const float* __restrict__ x, const float* __restrict__ w0,
                             const unsigned char* __restrict__ m0,
                             unsigned short* __restrict__ out,
                             const int* __restrict__ alist, const int* __restrict__ acnt,
                             int S, int outCS) {
    int i = blockIdx.x * blockDim.x + threadIdx.x;
    if (i >= *acnt) return;
    int vid = alist[i];
    int X = vid % S; int t1 = vid / S; int Y = t1 % S; int Z = t1 / S;
    long long Vv = (long long)S * S * S;
    float acc[16];
#pragma unroll
    for (int c = 0; c < 16; ++c) acc[c] = 0.f;
    for (int t = 0; t < 27; ++t) {
        int dz = t / 9 - 1, dy = (t / 3) % 3 - 1, dx = t % 3 - 1;
        int zz = Z + dz, yy = Y + dy, xx = X + dx;
        if ((unsigned)zz >= (unsigned)S || (unsigned)yy >= (unsigned)S ||
            (unsigned)xx >= (unsigned)S) continue;
        int nv = (zz * S + yy) * S + xx;
        if (!m0[nv]) continue;
        float i0 = x[nv], i1 = x[Vv + nv], i2 = x[2 * Vv + nv];
#pragma unroll
        for (int co = 0; co < 16; ++co) {
            const float* wp = w0 + (size_t)co * 81 + t;
            acc[co] += i0 * wp[0] + i1 * wp[27] + i2 * wp[54];
        }
    }
#pragma unroll
    for (int co = 0; co < 16; ++co) out[(size_t)vid * outCS + co] = f2bf(acc[co]);
}

// BN pass 1: per-block per-channel sum / sumsq (deterministic partials)
__global__ void bn_reduce(const unsigned short* __restrict__ buf, int stride, int C, int V,
                          float* __restrict__ part, int vPerBlk) {
    int c = threadIdx.x;
    if (c >= C) return;
    int v0 = blockIdx.x * vPerBlk;
    int v1 = v0 + vPerBlk; if (v1 > V) v1 = V;
    float s = 0.f, q = 0.f;
    for (int v = v0; v < v1; ++v) {
        float f = bf2f(buf[(size_t)v * stride + c]);
        s += f; q += f * f;
    }
    part[(size_t)blockIdx.x * 2 * C + c]     = s;
    part[(size_t)blockIdx.x * 2 * C + C + c] = q;
}

// BN pass 2: fold partials (fixed order) -> per-channel scale/shift
__global__ void bn_finalize(const float* __restrict__ part, int nBlk, int C,
                            const float* __restrict__ g, const float* __restrict__ b,
                            const int* __restrict__ acnt, float* __restrict__ ss) {
    int c = threadIdx.x;
    if (c >= C) return;
    float s = 0.f, q = 0.f;
    for (int i = 0; i < nBlk; ++i) {
        s += part[(size_t)i * 2 * C + c];
        q += part[(size_t)i * 2 * C + C + c];
    }
    int cnt = *acnt;
    float n = (float)(cnt > 0 ? cnt : 1);
    float mean = s / n;
    float var  = q / n - mean * mean;
    float sc = g[c] * rsqrtf(var + 1e-4f);
    ss[c]     = sc;
    ss[C + c] = b[c] - mean * sc;
}

// BN pass 3: apply scale/shift + ReLU over active voxels only
__global__ void bn_apply(const unsigned short* __restrict__ in, int inStride,
                         unsigned short* __restrict__ out, int outStride, int C,
                         const float* __restrict__ ss,
                         const int* __restrict__ alist, const int* __restrict__ acnt) {
    int gid = blockIdx.x * blockDim.x + threadIdx.x;
    int a = gid / C, c = gid % C;
    if (a >= *acnt) return;
    int v = alist[a];
    float f = bf2f(in[(size_t)v * inStride + c]);
    f = f * ss[c] + ss[C + c];
    f = fmaxf(f, 0.f);
    out[(size_t)v * outStride + c] = f2bf(f);
}

// Final BN+ReLU fused with 16->20 linear, scattered to dense NCDHW fp32 output
__global__ void out_linear(const unsigned short* __restrict__ h, int stride,
                           const float* __restrict__ ss, const float* __restrict__ wlin,
                           const float* __restrict__ blin, float* __restrict__ out,
                           const int* __restrict__ alist, const int* __restrict__ acnt, int V) {
    int i = blockIdx.x * blockDim.x + threadIdx.x;
    if (i >= *acnt) return;
    int v = alist[i];
    float hv[16];
#pragma unroll
    for (int c = 0; c < 16; ++c) {
        float f = bf2f(h[(size_t)v * stride + c]);
        hv[c] = fmaxf(f * ss[c] + ss[16 + c], 0.f);
    }
    for (int o = 0; o < 20; ++o) {
        float a = blin[o];
#pragma unroll
        for (int c = 0; c < 16; ++c) a += hv[c] * wlin[c * 20 + o];
        out[(size_t)o * V + v] = a;
    }
}

// ------------------------------- host --------------------------------------

extern "C" void kernel_launch(void* const* d_in, const int* in_sizes, int n_in,
                              void* d_out, int out_size, void* d_ws, size_t ws_size,
                              hipStream_t stream) {
    (void)in_sizes; (void)n_in; (void)ws_size;
    static const int PLv[7] = {16, 32, 48, 64, 80, 96, 112};
    int S[7], V[7], CS[7];
    for (int l = 0; l < 7; ++l) {
        S[l] = 128 >> l; V[l] = S[l] * S[l] * S[l];
        CS[l] = (l < 6) ? 2 * PLv[l] : PLv[l];
    }
    auto KPD = [](int c) { return (c + 31) & ~31; };

    // ---- workspace layout ----
    char* ws = (char*)d_ws;
    size_t off = 0;
    auto alloc = [&](size_t bytes) -> char* {
        off = (off + 255) & ~(size_t)255;
        char* p = ws + off; off += bytes; return p;
    };
    unsigned short *feat[7], *hin[7], *btmp[7];
    unsigned char* msk[7];
    int* alist[7];
    for (int l = 0; l < 7; ++l) feat[l] = (unsigned short*)alloc((size_t)V[l] * CS[l] * 2);
    for (int l = 0; l < 7; ++l) hin[l]  = (unsigned short*)alloc((size_t)V[l] * PLv[l] * 2);
    for (int l = 0; l < 7; ++l) btmp[l] = (unsigned short*)alloc((size_t)V[l] * CS[l] * 2);
    for (int l = 0; l < 7; ++l) msk[l]  = (unsigned char*)alloc((size_t)V[l]);
    for (int l = 0; l < 7; ++l) alist[l] = (int*)alloc((size_t)V[l] * 4);
    int* acnt = (int*)alloc(7 * 4);
    unsigned short *wb1t[7], *wb2t[6], *wdnt[6], *wupt[6];
    for (int l = 0; l < 7; ++l) wb1t[l] = (unsigned short*)alloc((size_t)27 * PLv[l] * KPD(PLv[l]) * 2);
    for (int l = 0; l < 6; ++l) wb2t[l] = (unsigned short*)alloc((size_t)27 * PLv[l] * KPD(2 * PLv[l]) * 2);
    for (int l = 0; l < 6; ++l) wdnt[l] = (unsigned short*)alloc((size_t)8 * PLv[l + 1] * KPD(PLv[l]) * 2);
    for (int l = 0; l < 6; ++l) wupt[l] = (unsigned short*)alloc((size_t)8 * PLv[l] * KPD(PLv[l + 1]) * 2);
    float* part = (float*)alloc((size_t)(1 << 20) * 4);
    float* ss   = (float*)alloc(512 * 4);
    size_t total = (off + 255) & ~(size_t)255;

    hipMemsetAsync(d_ws, 0, total, stream);                       // zero invariant + counters
    hipMemsetAsync(d_out, 0, (size_t)out_size * sizeof(float), stream);

    auto F = [&](int i) { return (const float*)d_in[i]; };
    const float* xin  = (const float*)d_in[0];
    const int*   mraw = (const int*)d_in[1];
    const float* w0   = (const float*)d_in[2];

    // ---- masks + active lists ----
    mask_init<<<(V[0] + 255) / 256, 256, 0, stream>>>(mraw, msk[0], V[0]);
    for (int l = 0; l < 6; ++l)
        mask_down<<<(V[l + 1] + 255) / 256, 256, 0, stream>>>(msk[l], msk[l + 1], S[l + 1]);
    for (int l = 0; l < 7; ++l)
        compact_kernel<<<(V[l] + 255) / 256, 256, 0, stream>>>(msk[l], alist[l], acnt + l, V[l]);

    // ---- weight transforms to bf16 [tap][Cout][Kpad] ----
    auto wx = [&](const float* src, unsigned short* dst, int Cout, int Cin, int nt, int coMajor) {
        int Kpad = KPD(Cin);
        long long tot = (long long)nt * Cout * Kpad;
        wxform<<<(int)((tot + 255) / 256), 256, 0, stream>>>(src, dst, Cout, Cin, nt, Kpad, coMajor);
    };
    for (int l = 0; l < 7; ++l) wx(F(3 + l),  wb1t[l], PLv[l],     PLv[l],     27, 1);
    for (int l = 0; l < 6; ++l) wx(F(10 + l), wb2t[l], PLv[l],     2 * PLv[l], 27, 1);
    for (int l = 0; l < 6; ++l) wx(F(16 + l), wdnt[l], PLv[l + 1], PLv[l],      8, 1);
    for (int l = 0; l < 6; ++l) wx(F(22 + l), wupt[l], PLv[l],     PLv[l + 1],  8, 0); // [ci][co][t]

    // ---- helper launchers ----
    auto bnrelu = [&](const unsigned short* inb, int inS, unsigned short* outb, int outS,
                      int C, int lvl, const float* g, const float* b) {
        int vpb = 4096;
        int nBlk = (V[lvl] + vpb - 1) / vpb;
        bn_reduce<<<nBlk, 256, 0, stream>>>(inb, inS, C, V[lvl], part, vpb);
        bn_finalize<<<1, 256, 0, stream>>>(part, nBlk, C, g, b, acnt + lvl, ss);
        long long tot = (long long)V[lvl] * C;
        bn_apply<<<(int)((tot + 255) / 256), 256, 0, stream>>>(inb, inS, outb, outS, C, ss,
                                                               alist[lvl], acnt + lvl);
    };
    auto spconv = [&](int mode, int ntaps, const unsigned short* inb, int inCS_,
                      const unsigned short* wtp, unsigned short* outb, int outCS_,
                      int outChOff, int Cin, int Cout, int lvl) {
        int Kpad = KPD(Cin);
        int nCoT = Cout / 16;
        int tiles = (V[lvl] + 15) / 16;
        int blocks = (tiles + 7) / 8;
        size_t smemB = (size_t)Cout * Kpad * 2;
        dim3 g(blocks), blk(256);
#define SPC(N) spconv_wmma<N><<<g, blk, smemB, stream>>>(inb, wtp, outb, alist[lvl], \
            acnt + lvl, S[lvl], S[lvl], S[lvl], Cin, Cout, Kpad, inCS_, outCS_,      \
            outChOff, mode, ntaps)
        switch (nCoT) {
            case 1: SPC(1); break;
            case 2: SPC(2); break;
            case 3: SPC(3); break;
            case 4: SPC(4); break;
            case 5: SPC(5); break;
            case 6: SPC(6); break;
            default: SPC(7); break;
        }
#undef SPC
    };

    // ---- stem ----
    conv0_kernel<<<(V[0] + 255) / 256, 256, 0, stream>>>(xin, w0, msk[0], hin[0],
                                                         alist[0], acnt + 0, S[0], PLv[0]);

    // ---- encoder ----
    for (int l = 0; l < 7; ++l) {
        bnrelu(hin[l], PLv[l], btmp[l], CS[l], PLv[l], l, F(28 + l), F(35 + l));          // bn_a
        spconv(MODE_SUB, 27, btmp[l], CS[l], wb1t[l], feat[l], CS[l], 0, PLv[l], PLv[l], l);
        if (l < 6) {
            bnrelu(feat[l], CS[l], btmp[l], CS[l], PLv[l], l, F(54 + l), F(60 + l));      // bn_b
            spconv(MODE_DOWN, 8, btmp[l], CS[l], wdnt[l], hin[l + 1], PLv[l + 1], 0,
                   PLv[l], PLv[l + 1], l + 1);
        }
    }
    // ---- decoder ----
    for (int l = 5; l >= 0; --l) {
        const unsigned short* z; int zs;
        if (l + 1 == 6) { z = feat[6]; zs = CS[6]; } else { z = hin[l + 1]; zs = PLv[l + 1]; }
        bnrelu(z, zs, btmp[l + 1], CS[l + 1], PLv[l + 1], l + 1, F(66 + l), F(72 + l));   // bn_c
        spconv(MODE_UP, 8, btmp[l + 1], CS[l + 1], wupt[l], feat[l], CS[l], PLv[l],
               PLv[l + 1], PLv[l], l);                        // concat into ch [p,2p)
        bnrelu(feat[l], CS[l], btmp[l], CS[l], 2 * PLv[l], l, F(42 + l), F(48 + l));      // bn_j
        spconv(MODE_SUB, 27, btmp[l], CS[l], wb2t[l], hin[l], PLv[l], 0,
               2 * PLv[l], PLv[l], l);
    }
    // ---- head: BN + ReLU + linear(16->20), masked scatter to dense output ----
    {
        int vpb = 4096, nBlk = (V[0] + vpb - 1) / vpb;
        bn_reduce<<<nBlk, 256, 0, stream>>>(hin[0], PLv[0], 16, V[0], part, vpb);
        bn_finalize<<<1, 256, 0, stream>>>(part, nBlk, 16, F(78), F(79), acnt + 0, ss);
        out_linear<<<(V[0] + 255) / 256, 256, 0, stream>>>(hin[0], PLv[0], ss, F(80), F(81),
                                                           (float*)d_out, alist[0], acnt + 0, V[0]);
    }
}
